// FCOSMaskPWPostProcessor_89154931130776
// MI455X (gfx1250) — matrix operand order repaired and checked by reading, not compiled, not run on hardware
//
#include <hip/hip_runtime.h>
#include <hip/hip_bf16.h>
#include <math.h>

typedef __attribute__((ext_vector_type(16))) _Float16 v16h;
typedef __attribute__((ext_vector_type(8)))  _Float16 v8h;
typedef __attribute__((ext_vector_type(8)))  float    v8f;

constexpr int NIMG = 2, C = 80, H = 32, W = 40, HW = H * W, M2 = 196;
constexpr int NFLAT = HW * C;        // 102400 candidates per image
constexpr int NSORT = 131072;        // next pow2 for bitonic sort
constexpr int KTOP  = 1000;          // PRE_NMS_TOP_N
constexpr int KBOX  = 1024;          // padded to 64 WMMA M-tiles
constexpr int KD    = H * W;         // 1280 = GEMM K dimension (i*W+j)
constexpr int CPAD  = 208;           // 196 channels -> 13 WMMA N-tiles
constexpr int KOUT  = 100;           // FPN_POST_NMS_TOP_N
constexpr int MAXH  = 512, MAXW = 512;
constexpr float PRE_NMS_THRESH = 0.05f, NMS_THRESH = 0.6f;
constexpr float CLS_OFFSET = 10000.0f, NEGV = -1e9f;

// ---- workspace layout (~10.9 MB total) ----
constexpr size_t AL(size_t x) { return (x + 255) & ~size_t(255); }
constexpr size_t OFF_KEYS = 0;
constexpr size_t OFF_WR   = AL(OFF_KEYS + sizeof(unsigned long long) * NSORT * NIMG);
constexpr size_t OFF_WC   = AL(OFF_WR   + sizeof(float) * (MAXH + 1) * H);
constexpr size_t OFF_BX   = AL(OFF_WC   + sizeof(float) * (MAXW + 1) * W);
constexpr size_t OFF_SC   = AL(OFF_BX   + sizeof(float) * NIMG * KBOX * 4);
constexpr size_t OFF_LB   = AL(OFF_SC   + sizeof(float) * NIMG * KBOX);
constexpr size_t OFF_VAL  = AL(OFF_LB   + sizeof(float) * NIMG * KBOX);
constexpr size_t OFF_OK   = AL(OFF_VAL  + sizeof(float) * NIMG * KBOX);
constexpr size_t OFF_AREA = AL(OFF_OK   + sizeof(float) * NIMG * KBOX);
constexpr size_t OFF_AB   = AL(OFF_AREA + sizeof(float) * NIMG * KBOX);
constexpr size_t OFF_AM   = AL(OFF_AB   + sizeof(float) * NIMG * KBOX * (H + W));
constexpr size_t OFF_MT   = AL(OFF_AM   + sizeof(_Float16) * NIMG * KBOX * KD);
constexpr size_t OFF_S    = AL(OFF_MT   + sizeof(_Float16) * NIMG * CPAD * KD);
constexpr size_t OFF_KEEP = AL(OFF_S    + sizeof(float) * NIMG * KBOX * CPAD);

__device__ __forceinline__ int imin(int a, int b) { return a < b ? a : b; }
__device__ __forceinline__ int imax(int a, int b) { return a > b ? a : b; }
__device__ __forceinline__ float sigmoidf(float x) { return 1.0f / (1.0f + __expf(-x)); }

// ---------- 1. build sortable keys: (ordered-float << 32) | ~idx ----------
__global__ void k_keys(const float* __restrict__ cls, const float* __restrict__ cent,
                       unsigned long long* __restrict__ keys) {
  int gid = blockIdx.x * blockDim.x + threadIdx.x;
  int n = gid / NSORT, t = gid - n * NSORT;
  if (n >= NIMG) return;
  unsigned long long kv = 0ull;  // padding: sorts below every real entry
  if (t < NFLAT) {
    int loc = t / C, c = t - loc * C;
    float s = sigmoidf(cls[(n * C + c) * HW + loc]);
    float val = NEGV;
    if (s > PRE_NMS_THRESH) val = s * sigmoidf(cent[n * HW + loc]);
    unsigned ub = __float_as_uint(val);
    unsigned uk = (ub & 0x80000000u) ? ~ub : (ub | 0x80000000u);  // order-preserving
    kv = ((unsigned long long)uk << 32) | (unsigned)(~(unsigned)t);  // ties: low idx first
  }
  keys[(size_t)n * NSORT + t] = kv;
}

// ---------- 2. one bitonic compare-exchange pass (descending) ----------
__global__ void k_bitonic(unsigned long long* __restrict__ keys, int k, int j) {
  unsigned long long* d = keys + (size_t)blockIdx.y * NSORT;
  int i = blockIdx.x * blockDim.x + threadIdx.x;
  int ixj = i ^ j;
  if (ixj > i) {
    unsigned long long a = d[i], b = d[ixj];
    bool up = ((i & k) == 0);
    if (up ? (a < b) : (a > b)) { d[i] = b; d[ixj] = a; }
  }
}

// ---------- 3. prefix-sum bilinear weight tables Wr[513][32], Wc[513][40] ----------
__global__ void k_tables(const int* __restrict__ hp, const int* __restrict__ wp, char* ws) {
  float* Wr = (float*)(ws + OFF_WR);
  float* Wc = (float*)(ws + OFF_WC);
  int h = imin(*hp, MAXH), w = imin(*wp, MAXW);
  int tid = threadIdx.x;
  if (tid < H) {
    int i = tid; float acc = 0.f; Wr[i] = 0.f;
    float scale = (float)H / (float)h;
    for (int r = 0; r < h; ++r) {
      float src = ((float)r + 0.5f) * scale - 0.5f;
      src = fminf(fmaxf(src, 0.f), (float)(H - 1));
      int i0 = (int)floorf(src); int i1 = imin(i0 + 1, H - 1);
      float lam = src - (float)i0;
      acc += (i == i0 ? (1.f - lam) : 0.f) + (i == i1 ? lam : 0.f);
      Wr[(r + 1) * H + i] = acc;
    }
  } else if (tid >= 64 && tid < 64 + W) {
    int j = tid - 64; float acc = 0.f; Wc[j] = 0.f;
    float scale = (float)W / (float)w;
    for (int r = 0; r < w; ++r) {
      float src = ((float)r + 0.5f) * scale - 0.5f;
      src = fminf(fmaxf(src, 0.f), (float)(W - 1));
      int j0 = (int)floorf(src); int j1 = imin(j0 + 1, W - 1);
      float lam = src - (float)j0;
      acc += (j == j0 ? (1.f - lam) : 0.f) + (j == j1 ? lam : 0.f);
      Wc[(r + 1) * W + j] = acc;
    }
  }
}

// ---------- 4. decode top-K boxes + per-box corner-difference weight vectors ----------
__global__ void k_decode(const float* __restrict__ locations, const float* __restrict__ reg,
                         const int* __restrict__ hp, const int* __restrict__ wp, char* ws) {
  int n = blockIdx.y;
  int k = blockIdx.x * blockDim.x + threadIdx.x;
  if (k >= KBOX) return;
  const unsigned long long* keys = (const unsigned long long*)(ws + OFF_KEYS) + (size_t)n * NSORT;
  float* bx = (float*)(ws + OFF_BX) + ((size_t)n * KBOX + k) * 4;
  float* sc = (float*)(ws + OFF_SC) + n * KBOX;
  float* lb = (float*)(ws + OFF_LB) + n * KBOX;
  float* vl = (float*)(ws + OFF_VAL) + n * KBOX;
  float* ok = (float*)(ws + OFF_OK) + n * KBOX;
  float* ar = (float*)(ws + OFF_AREA) + n * KBOX;
  float* ab = (float*)(ws + OFF_AB) + ((size_t)n * KBOX + k) * (H + W);
  const float* Wr = (const float*)(ws + OFF_WR);
  const float* Wc = (const float*)(ws + OFF_WC);
  int h = imin(*hp, MAXH), w = imin(*wp, MAXW);

  bool valid = false; float val = 0.f; unsigned idx = 0;
  if (k < KTOP) {
    unsigned long long kv = keys[k];
    unsigned uk = (unsigned)(kv >> 32);
    unsigned ub = (uk & 0x80000000u) ? (uk & 0x7FFFFFFFu) : ~uk;
    val = __uint_as_float(ub);
    idx = ~(unsigned)(kv & 0xFFFFFFFFu);
    valid = (val > NEGV * 0.5f) && (idx < (unsigned)NFLAT);
  }
  if (!valid) {  // sanitized padding: zero weights -> S row = 0, prob = 0
    bx[0] = bx[1] = bx[2] = bx[3] = 0.f;
    sc[k] = 0.f; lb[k] = 1.f; vl[k] = 0.f; ok[k] = 0.f; ar[k] = 1.f;
    for (int i = 0; i < H + W; ++i) ab[i] = 0.f;
    return;
  }
  int loc = idx / C, cc = idx - loc * C;
  float lx = locations[loc * 2 + 0], ly = locations[loc * 2 + 1];
  float b0 = lx - reg[(n * 4 + 0) * HW + loc];
  float b1 = ly - reg[(n * 4 + 1) * HW + loc];
  float b2 = lx + reg[(n * 4 + 2) * HW + loc];
  float b3 = ly + reg[(n * 4 + 3) * HW + loc];
  // mask region: truncate toward zero (.int()), h/w swap replicated from reference
  int x1 = imax((int)b0, 0), x2 = imin((int)b2, h - 1);
  int y1 = imax((int)b1, 0), y2 = imin((int)b3, w - 1);
  bool rok = (x1 < x2) && (y1 < x2) && (y1 < y2);  // replicated (y1 < x2) bug
  int x1c = imin(imax(x1, 0), h), x2c = imin(imax(x2, 0), h);
  int y1c = imin(imax(y1, 0), w), y2c = imin(imax(y2, 0), w);
  float area = (float)imax((x2c - x1c) * (y2c - y1c), 1);
  // clip_to_image: x vs w-1, y vs h-1
  bx[0] = fminf(fmaxf(b0, 0.f), (float)(w - 1));
  bx[1] = fminf(fmaxf(b1, 0.f), (float)(h - 1));
  bx[2] = fminf(fmaxf(b2, 0.f), (float)(w - 1));
  bx[3] = fminf(fmaxf(b3, 0.f), (float)(h - 1));
  sc[k] = sqrtf(val); lb[k] = (float)(cc + 1); vl[k] = 1.f;
  ok[k] = rok ? 1.f : 0.f; ar[k] = area;
  for (int i = 0; i < H; ++i) ab[i]     = Wr[x2c * H + i] - Wr[x1c * H + i];
  for (int j = 0; j < W; ++j) ab[H + j] = Wc[y2c * W + j] - Wc[y1c * W + j];
}

// ---------- 5. B matrix: sigmoid(mask) transposed to [CPAD][KD] f16 ----------
__global__ void k_mprep(const float* __restrict__ mask, char* ws) {
  int gid = blockIdx.x * blockDim.x + threadIdx.x;
  int n = gid / (CPAD * KD), r = gid - n * (CPAD * KD);
  int c = r / KD, k = r - c * KD;
  if (n >= NIMG) return;
  float v = 0.f;
  if (c < M2) v = sigmoidf(mask[(size_t)n * M2 * KD + (size_t)c * KD + k]);
  ((_Float16*)(ws + OFF_MT))[gid] = (_Float16)v;
}

// ---------- 6. A matrix: per-box outer product a_i * b_j, f16 [KBOX][KD] ----------
__global__ void k_amat(char* ws) {
  int n = blockIdx.y, k = blockIdx.x;
  const float* ab = (const float*)(ws + OFF_AB) + ((size_t)n * KBOX + k) * (H + W);
  _Float16* A = (_Float16*)(ws + OFF_AM) + ((size_t)n * KBOX + k) * KD;
  __shared__ float sa[H], sb[W];
  int t = threadIdx.x;
  if (t < H) sa[t] = ab[t];
  if (t >= H && t < H + W) sb[t - H] = ab[t];
  __syncthreads();
  for (int x = t; x < KD; x += blockDim.x) {
    int i = x / W, j = x - i * W;
    A[x] = (_Float16)(sa[i] * sb[j]);
  }
}

// ---------- 7. WMMA GEMM: S[1024x208] = A[1024x1280] x Mt^T, f16 in / f32 acc ----------
__global__ void k_gemm(char* ws) {
  int n = blockIdx.z;
  const _Float16* A = (const _Float16*)(ws + OFF_AM) + (size_t)n * KBOX * KD;
  const _Float16* B = (const _Float16*)(ws + OFF_MT) + (size_t)n * CPAD * KD;
  float* S = (float*)(ws + OFF_S) + (size_t)n * KBOX * CPAD;
  int lane = threadIdx.x;
  int hi = lane >> 4;                       // half-wave select
  const _Float16* arow = A + (size_t)(blockIdx.x * 16 + (lane & 15)) * KD;
  const _Float16* brow = B + (size_t)(blockIdx.y * 16 + (lane & 15)) * KD;
  // A layout (matches B): lanes0-15 hold K 0-15, lanes16-31 hold K 16-31
  int aoff0 = hi * 16, aoff1 = hi * 16 + 8;
  int boff = hi * 16;                       // B col-major: lanes0-15 K0-15, lanes16-31 K16-31
  v8f acc = {};
  for (int kk = 0; kk < KD; kk += 32) {
    union { v16h v; v8h h[2]; } af, bf;
    af.h[0] = *(const v8h*)(arow + kk + aoff0);
    af.h[1] = *(const v8h*)(arow + kk + aoff1);
    bf.h[0] = *(const v8h*)(brow + kk + boff);
    bf.h[1] = *(const v8h*)(brow + kk + boff + 8);
    if (kk + 64 < KD) {
      __builtin_prefetch(arow + kk + 64, 0, 1);   // global_prefetch_b8
      __builtin_prefetch(brow + kk + 64, 0, 1);
    }
    acc = __builtin_amdgcn_wmma_f32_16x16x32_f16(false, af.v, false, bf.v,
                                                 (short)0, acc, false, false);
  }
  int col = blockIdx.y * 16 + (lane & 15);
  int rbase = blockIdx.x * 16 + hi * 8;     // D: VGPR d -> M = d (+8 for lanes 16-31)
#pragma unroll
  for (int d = 0; d < 8; ++d) S[(size_t)(rbase + d) * CPAD + col] = acc[d];
}

// ---------- 8. sequential greedy per-class NMS (boxes+class offset in LDS) ----------
__global__ void k_nms(char* ws) {
  int n = blockIdx.x, t = threadIdx.x;
  __shared__ float nb[KTOP][4];
  __shared__ unsigned char svalid[KTOP], skeep[KTOP];
  const float* bx = (const float*)(ws + OFF_BX) + (size_t)n * KBOX * 4;
  const float* lb = (const float*)(ws + OFF_LB) + n * KBOX;
  const float* vl = (const float*)(ws + OFF_VAL) + n * KBOX;
  unsigned char* keep = (unsigned char*)(ws + OFF_KEEP) + n * KBOX;
  if (t < KTOP) {
    float off = lb[t] * CLS_OFFSET;
    nb[t][0] = bx[t * 4 + 0] + off; nb[t][1] = bx[t * 4 + 1] + off;
    nb[t][2] = bx[t * 4 + 2] + off; nb[t][3] = bx[t * 4 + 3] + off;
    svalid[t] = vl[t] > 0.5f; skeep[t] = 1;
  }
  __syncthreads();
  for (int i = 0; i < KTOP; ++i) {
    bool ki = skeep[i] && svalid[i];
    if (ki && t > i && t < KTOP && skeep[t]) {
      float iw = fminf(nb[i][2], nb[t][2]) - fmaxf(nb[i][0], nb[t][0]) + 1.f;
      float ih = fminf(nb[i][3], nb[t][3]) - fmaxf(nb[i][1], nb[t][1]) + 1.f;
      float inter = fmaxf(iw, 0.f) * fmaxf(ih, 0.f);
      float areaA = (nb[i][2] - nb[i][0] + 1.f) * (nb[i][3] - nb[i][1] + 1.f);
      float areaB = (nb[t][2] - nb[t][0] + 1.f) * (nb[t][3] - nb[t][1] + 1.f);
      if (inter / fmaxf(areaA + areaB - inter, 1e-6f) > NMS_THRESH) skeep[t] = 0;
    }
    __syncthreads();
  }
  if (t < KTOP) keep[t] = (unsigned char)(skeep[t] && svalid[t]);
  else if (t < KBOX) keep[t] = 0;
}

// ---------- 9. final top-100 gather (kept-first order == top_k tie semantics) ----------
__global__ void k_final(char* ws, float* __restrict__ out) {
  int n = blockIdx.x, t = threadIdx.x;
  __shared__ int sel[KOUT];
  __shared__ float svf[KOUT];
  const unsigned char* keep = (const unsigned char*)(ws + OFF_KEEP) + n * KBOX;
  if (t == 0) {
    int cnt = 0;
    for (int i = 0; i < KTOP && cnt < KOUT; ++i) if (keep[i]) { sel[cnt] = i; svf[cnt] = 1.f; cnt++; }
    for (int i = 0; i < KTOP && cnt < KOUT; ++i) if (!keep[i]) { sel[cnt] = i; svf[cnt] = 0.f; cnt++; }
    for (; cnt < KOUT; ++cnt) { sel[cnt] = 0; svf[cnt] = 0.f; }
  }
  __syncthreads();
  const float* bx = (const float*)(ws + OFF_BX) + (size_t)n * KBOX * 4;
  const float* sc = (const float*)(ws + OFF_SC) + n * KBOX;
  const float* lb = (const float*)(ws + OFF_LB) + n * KBOX;
  const float* ok = (const float*)(ws + OFF_OK) + n * KBOX;
  const float* ar = (const float*)(ws + OFF_AREA) + n * KBOX;
  const float* S  = (const float*)(ws + OFF_S) + (size_t)n * KBOX * CPAD;
  float* obx = out + (size_t)n * KOUT * 4;
  float* osc = out + NIMG * KOUT * 4 + n * KOUT;
  float* olb = out + NIMG * KOUT * 5 + n * KOUT;
  float* opr = out + NIMG * KOUT * 6 + (size_t)n * KOUT * M2;
  float* ovf = out + NIMG * KOUT * 6 + (size_t)NIMG * KOUT * M2 + n * KOUT;
  if (t < KOUT) {
    int k = sel[t]; float v = svf[t];
    obx[t * 4 + 0] = bx[k * 4 + 0]; obx[t * 4 + 1] = bx[k * 4 + 1];
    obx[t * 4 + 2] = bx[k * 4 + 2]; obx[t * 4 + 3] = bx[k * 4 + 3];
    osc[t] = sc[k] * v; olb[t] = lb[k]; ovf[t] = v;
  }
  for (int x = t; x < KOUT * M2; x += blockDim.x) {
    int s = x / M2, c = x - s * M2;
    int k = sel[s];
    opr[x] = (S[(size_t)k * CPAD + c] / ar[k]) * ok[k] * svf[s];
  }
}

extern "C" void kernel_launch(void* const* d_in, const int* in_sizes, int n_in,
                              void* d_out, int out_size, void* d_ws, size_t ws_size,
                              hipStream_t stream) {
  (void)in_sizes; (void)n_in; (void)out_size; (void)ws_size;
  const float* locations  = (const float*)d_in[0];
  const float* box_cls    = (const float*)d_in[1];
  const float* box_reg    = (const float*)d_in[2];
  const float* centerness = (const float*)d_in[3];
  const float* box_mask   = (const float*)d_in[4];
  const int*   hp         = (const int*)d_in[5];
  const int*   wp         = (const int*)d_in[6];
  char* ws = (char*)d_ws;
  float* out = (float*)d_out;

  k_keys<<<(NIMG * NSORT) / 256, 256, 0, stream>>>(
      box_cls, centerness, (unsigned long long*)(ws + OFF_KEYS));
  for (int k = 2; k <= NSORT; k <<= 1)
    for (int j = k >> 1; j > 0; j >>= 1)
      k_bitonic<<<dim3(NSORT / 256, NIMG), 256, 0, stream>>>(
          (unsigned long long*)(ws + OFF_KEYS), k, j);
  k_tables<<<1, 128, 0, stream>>>(hp, wp, ws);
  k_decode<<<dim3(KBOX / 128, NIMG), 128, 0, stream>>>(locations, box_reg, hp, wp, ws);
  k_mprep<<<(NIMG * CPAD * KD) / 256, 256, 0, stream>>>(box_mask, ws);
  k_amat<<<dim3(KBOX, NIMG), 128, 0, stream>>>(ws);
  k_gemm<<<dim3(KBOX / 16, CPAD / 16, NIMG), 32, 0, stream>>>(ws);
  k_nms<<<NIMG, 1024, 0, stream>>>(ws);
  k_final<<<NIMG, 1024, 0, stream>>>(ws, out);
}
